// _PointnetSAModuleBase_48163763257797
// MI455X (gfx1250) — matrix operand order repaired and checked by reading, not compiled, not run on hardware
//
#include <hip/hip_runtime.h>

typedef __attribute__((ext_vector_type(16))) _Float16 v16h;
typedef __attribute__((ext_vector_type(8)))  _Float16 v8h;
typedef __attribute__((ext_vector_type(8)))  float    v8f;

#define NB      4
#define NPTS    8192
#define NPOINT  2048
#define NSAMPLE 32
#define CIN     64
#define R2      0.01f   // 0.1^2

// ---------------------------------------------------------------------------
// Kernel 1: farthest point sampling. One block per batch, 512 threads,
// each thread owns 16 points (interleaved for coalescing). Writes new_xyz
// (the centroid entering each step, matching the reference scan) to d_out.
// ---------------------------------------------------------------------------
__global__ __launch_bounds__(512) void fps_kernel(const float* __restrict__ xyz,
                                                  float* __restrict__ newxyz)
{
    const int b    = blockIdx.x;
    const int tid  = threadIdx.x;
    const int lane = tid & 31;
    const int w    = tid >> 5;              // 16 waves
    const float* P = xyz + (size_t)b * NPTS * 3;

    float px[16], py[16], pz[16], dist[16];
#pragma unroll
    for (int j = 0; j < 16; ++j) {
        int p = j * 512 + tid;
        px[j] = P[p * 3 + 0]; py[j] = P[p * 3 + 1]; pz[j] = P[p * 3 + 2];
        dist[j] = 1e10f;
    }

    __shared__ float sc[3];
    __shared__ float sv[16];
    __shared__ int   si[16];
    __shared__ int   sFar;

    int far = 0;
    for (int i = 0; i < NPOINT; ++i) {
        if (tid == 0) {
            float cx = P[far * 3 + 0], cy = P[far * 3 + 1], cz = P[far * 3 + 2];
            sc[0] = cx; sc[1] = cy; sc[2] = cz;
            float* o = newxyz + ((size_t)b * NPOINT + i) * 3;
            o[0] = cx; o[1] = cy; o[2] = cz;
        }
        __syncthreads();
        float cx = sc[0], cy = sc[1], cz = sc[2];

        float best = -1.0f; int bi = 0x7fffffff;
#pragma unroll
        for (int j = 0; j < 16; ++j) {
            float dx = px[j] - cx, dy = py[j] - cy, dz = pz[j] - cz;
            float d  = dx * dx + dy * dy + dz * dz;
            dist[j]  = dist[j] < d ? dist[j] : d;
            int p = j * 512 + tid;
            if (dist[j] > best || (dist[j] == best && p < bi)) { best = dist[j]; bi = p; }
        }
        // wave32 argmax reduce
        for (int d = 1; d < 32; d <<= 1) {
            float ov = __shfl_xor(best, d, 32);
            int   oi = __shfl_xor(bi,   d, 32);
            if (ov > best || (ov == best && oi < bi)) { best = ov; bi = oi; }
        }
        if (lane == 0) { sv[w] = best; si[w] = bi; }
        __syncthreads();
        if (w == 0) {
            float v  = lane < 16 ? sv[lane] : -1e30f;
            int   ix = lane < 16 ? si[lane] : 0x7fffffff;
            for (int d = 1; d < 32; d <<= 1) {
                float ov = __shfl_xor(v,  d, 32);
                int   oi = __shfl_xor(ix, d, 32);
                if (ov > v || (ov == v && oi < ix)) { v = ov; ix = oi; }
            }
            if (lane == 0) sFar = ix;
        }
        __syncthreads();
        far = sFar;
    }
}

// ---------------------------------------------------------------------------
// Kernel 2: weight conversion. Pack W0 (64x67 f32) into f16 [64][96] with
// channel reorder (features first, xyz at 64..66, zero pad 67..95); W1/W2
// straight f32->f16 copies.
// ---------------------------------------------------------------------------
__global__ void convert_weights(const float* __restrict__ W0,
                                const float* __restrict__ W1,
                                const float* __restrict__ W2,
                                _Float16* __restrict__ W0h,
                                _Float16* __restrict__ W1h,
                                _Float16* __restrict__ W2h)
{
    int t = blockIdx.x * blockDim.x + threadIdx.x;
    if (t < 64 * 96) {
        int o = t / 96, c = t % 96;
        float v;
        if (c < 64)      v = W0[o * 67 + 3 + c];      // feature channels
        else if (c < 67) v = W0[o * 67 + (c - 64)];   // xyz channels
        else             v = 0.0f;                    // pad
        W0h[t] = (_Float16)v;
    } else if (t < 64 * 96 + 64 * 64) {
        int u = t - 64 * 96;
        W1h[u] = (_Float16)W1[u];
    } else if (t < 64 * 96 + 64 * 64 + 128 * 64) {
        int u = t - 64 * 96 - 64 * 64;
        W2h[u] = (_Float16)W2[u];
    }
}

// ---------------------------------------------------------------------------
// Kernel 3: features (B,64,N) f32 -> (B,N,64) f16 (pixel-major for gather).
// ---------------------------------------------------------------------------
__global__ void convert_features(const float* __restrict__ feat,
                                 _Float16* __restrict__ fth)
{
    int t = blockIdx.x * blockDim.x + threadIdx.x;   // (b,c,n), n fastest
    int n = t & (NPTS - 1);
    int c = (t >> 13) & 63;
    int b = t >> 19;
    fth[((size_t)b * NPTS + n) * CIN + c] = (_Float16)feat[t];
}

// ---------------------------------------------------------------------------
// Kernel 4: ball query. One wave per center; ballot + prefix popcount keeps
// the reference's lowest-index-first selection; fill with first index.
// ---------------------------------------------------------------------------
__global__ __launch_bounds__(256) void ball_query_kernel(const float* __restrict__ xyz,
                                                         const float* __restrict__ newxyz,
                                                         int* __restrict__ ballIdx)
{
    const int lane = threadIdx.x & 31;
    const int cg   = blockIdx.x * 8 + (threadIdx.x >> 5);   // global center
    const int b    = cg >> 11;                              // /2048
    const float* P = xyz + (size_t)b * NPTS * 3;

    float cx = newxyz[cg * 3 + 0];
    float cy = newxyz[cg * 3 + 1];
    float cz = newxyz[cg * 3 + 2];

    int cnt = 0, first = 0;
    for (int base = 0; base < NPTS; base += 32) {
        int j = base + lane;
        float dx = P[j * 3 + 0] - cx;
        float dy = P[j * 3 + 1] - cy;
        float dz = P[j * 3 + 2] - cz;
        bool pred = (dx * dx + dy * dy + dz * dz) <= R2;
        unsigned mask = __builtin_amdgcn_ballot_w32(pred);
        if (cnt == 0 && mask) first = base + __builtin_ctz(mask);
        int prefix = __builtin_popcount(mask & ((1u << lane) - 1u));
        int slot = cnt + prefix;
        if (pred && slot < NSAMPLE) ballIdx[(size_t)cg * NSAMPLE + slot] = j;
        cnt += __builtin_popcount(mask);
        if (cnt >= NSAMPLE) break;
    }
    if (cnt < NSAMPLE && lane >= cnt)
        ballIdx[(size_t)cg * NSAMPLE + lane] = first;   // first==0 if none found
}

// ---------------------------------------------------------------------------
// Kernel 5: fused gather + 3-layer WMMA MLP + max-over-samples.
// Block = 128 threads (4 waves), 8 centers per block (2 per wave, 64 pixels).
// LDS: X0 [256][96] f16 (input / layer1 out), X1 [256][64] f16 (layer0 out).
// ---------------------------------------------------------------------------
__device__ __forceinline__ v16h load_afrag(const _Float16* wp) {
    v8h lo = *(const v8h*)(wp);        // K = ktile*32 + half*8 + 0..7
    v8h hi = *(const v8h*)(wp + 16);   // K = ktile*32 + 16 + half*8 + 0..7
    v16h a;
#pragma unroll
    for (int e = 0; e < 8; ++e) { a[e] = lo[e]; a[e + 8] = hi[e]; }
    return a;
}

__global__ __launch_bounds__(128) void mlp_fused_kernel(
    const int* __restrict__ ballIdx, const _Float16* __restrict__ fth,
    const float* __restrict__ xyz,   const float* __restrict__ newxyz,
    const _Float16* __restrict__ W0h, const _Float16* __restrict__ W1h,
    const _Float16* __restrict__ W2h,
    const float* __restrict__ b0, const float* __restrict__ b1,
    const float* __restrict__ b2,
    float* __restrict__ outFeat)
{
    __shared__ _Float16 X0[256 * 96];   // 49152 B
    __shared__ _Float16 X1[256 * 64];   // 32768 B

    const int tid  = threadIdx.x;
    const int lane = tid & 31;
    const int w    = tid >> 5;            // wave 0..3
    const int b    = blockIdx.x >> 8;     // 256 tiles per batch
    const int tile = blockIdx.x & 255;
    const int centerBase = tile * 8 + w * 2;   // in-batch center base (2/wave)
    const int pixBase = w * 64;                // block-local pixel base
    const int half = lane >> 4;
    const int n    = lane & 15;

    // ---- gather: features (8 lanes/pixel, v8h) + xyz tail channels ----
    {
        const int g = lane >> 3, e = lane & 7;
        for (int pp = 0; pp < 16; ++pp) {
            int p  = pp * 4 + g;                    // 0..63 within wave
            int cg = b * NPOINT + centerBase + (p >> 5);
            int idx = ballIdx[(size_t)cg * NSAMPLE + (p & 31)];
            v8h v = *(const v8h*)(fth + ((size_t)b * NPTS + idx) * CIN + e * 8);
            *(v8h*)(&X0[(pixBase + p) * 96 + e * 8]) = v;
        }
        for (int it = 0; it < 2; ++it) {
            int p  = it * 32 + lane;
            int cg = b * NPOINT + centerBase + (p >> 5);
            int idx = ballIdx[(size_t)cg * NSAMPLE + (p & 31)];
            float cx = newxyz[cg * 3 + 0], cy = newxyz[cg * 3 + 1], cz = newxyz[cg * 3 + 2];
            const float* q = xyz + ((size_t)b * NPTS + idx) * 3;
            _Float16* row = &X0[(pixBase + p) * 96];
            row[64] = (_Float16)(q[0] - cx);
            row[65] = (_Float16)(q[1] - cy);
            row[66] = (_Float16)(q[2] - cz);
#pragma unroll
            for (int c = 67; c < 96; ++c) row[c] = (_Float16)0.0f;
        }
    }
    __syncthreads();

    // ---- layer 0: 96 -> 64 ----
    for (int nt = 0; nt < 4; ++nt) {
        const _Float16* bp = &X0[(pixBase + nt * 16 + n) * 96 + half * 8];
        v16h bf0 = load_afrag(bp);
        v16h bf1 = load_afrag(bp + 32);
        v16h bf2 = load_afrag(bp + 64);
#pragma unroll
        for (int mt = 0; mt < 4; ++mt) {
            const int m = mt * 16 + n;
            v8f acc = {};
            acc = __builtin_amdgcn_wmma_f32_16x16x32_f16(false, load_afrag(W0h + m * 96 +  0 + half * 8), false, bf0, (short)0, acc, false, false);
            acc = __builtin_amdgcn_wmma_f32_16x16x32_f16(false, load_afrag(W0h + m * 96 + 32 + half * 8), false, bf1, (short)0, acc, false, false);
            acc = __builtin_amdgcn_wmma_f32_16x16x32_f16(false, load_afrag(W0h + m * 96 + 64 + half * 8), false, bf2, (short)0, acc, false, false);
            const float* bb = b0 + mt * 16 + half * 8;
            v8h oh;
#pragma unroll
            for (int r = 0; r < 8; ++r) {
                float v = acc[r] + bb[r];
                oh[r] = (_Float16)(v > 0.0f ? v : 0.0f);
            }
            *(v8h*)(&X1[(pixBase + nt * 16 + n) * 64 + mt * 16 + half * 8]) = oh;
        }
    }

    // ---- layer 1: 64 -> 64 (output into X0 region, stride 96) ----
    for (int nt = 0; nt < 4; ++nt) {
        const _Float16* bp = &X1[(pixBase + nt * 16 + n) * 64 + half * 8];
        v16h bf0 = load_afrag(bp);
        v16h bf1 = load_afrag(bp + 32);
#pragma unroll
        for (int mt = 0; mt < 4; ++mt) {
            const int m = mt * 16 + n;
            v8f acc = {};
            acc = __builtin_amdgcn_wmma_f32_16x16x32_f16(false, load_afrag(W1h + m * 64 +  0 + half * 8), false, bf0, (short)0, acc, false, false);
            acc = __builtin_amdgcn_wmma_f32_16x16x32_f16(false, load_afrag(W1h + m * 64 + 32 + half * 8), false, bf1, (short)0, acc, false, false);
            const float* bb = b1 + mt * 16 + half * 8;
            v8h oh;
#pragma unroll
            for (int r = 0; r < 8; ++r) {
                float v = acc[r] + bb[r];
                oh[r] = (_Float16)(v > 0.0f ? v : 0.0f);
            }
            *(v8h*)(&X0[(pixBase + nt * 16 + n) * 96 + mt * 16 + half * 8]) = oh;
        }
    }

    // ---- layer 2: 64 -> 128 + max over 32 samples ----
    for (int ci = 0; ci < 2; ++ci) {
        const _Float16* r0 = &X0[(pixBase + ci * 32 +      n) * 96 + half * 8];
        const _Float16* r1 = &X0[(pixBase + ci * 32 + 16 + n) * 96 + half * 8];
        v16h b00 = load_afrag(r0);
        v16h b01 = load_afrag(r0 + 32);
        v16h b10 = load_afrag(r1);
        v16h b11 = load_afrag(r1 + 32);
        const int center = centerBase + ci;
#pragma unroll
        for (int mt = 0; mt < 8; ++mt) {
            const int m = mt * 16 + n;
            v16h af0 = load_afrag(W2h + m * 64 +  0 + half * 8);
            v16h af1 = load_afrag(W2h + m * 64 + 32 + half * 8);
            v8f a0 = {}, a1 = {};
            a0 = __builtin_amdgcn_wmma_f32_16x16x32_f16(false, af0, false, b00, (short)0, a0, false, false);
            a1 = __builtin_amdgcn_wmma_f32_16x16x32_f16(false, af0, false, b10, (short)0, a1, false, false);
            a0 = __builtin_amdgcn_wmma_f32_16x16x32_f16(false, af1, false, b01, (short)0, a0, false, false);
            a1 = __builtin_amdgcn_wmma_f32_16x16x32_f16(false, af1, false, b11, (short)0, a1, false, false);
            const float* bb = b2 + mt * 16 + half * 8;
            float mx[8];
#pragma unroll
            for (int r = 0; r < 8; ++r) {
                float v0 = a0[r] + bb[r]; v0 = v0 > 0.0f ? v0 : 0.0f;
                float v1 = a1[r] + bb[r]; v1 = v1 > 0.0f ? v1 : 0.0f;
                float v = v0 > v1 ? v0 : v1;
                for (int d = 1; d < 16; d <<= 1) {
                    float o = __shfl_xor(v, d, 32);
                    v = o > v ? o : v;
                }
                mx[r] = v;
            }
            if (n == 0) {
#pragma unroll
                for (int r = 0; r < 8; ++r)
                    outFeat[((size_t)b * 128 + mt * 16 + half * 8 + r) * NPOINT + center] = mx[r];
            }
        }
    }
}

// ---------------------------------------------------------------------------
extern "C" void kernel_launch(void* const* d_in, const int* in_sizes, int n_in,
                              void* d_out, int out_size, void* d_ws, size_t ws_size,
                              hipStream_t stream) {
    (void)in_sizes; (void)n_in; (void)out_size; (void)ws_size;
    const float* xyz  = (const float*)d_in[0];
    const float* feat = (const float*)d_in[1];
    const float* W0   = (const float*)d_in[2];
    const float* b0   = (const float*)d_in[3];
    const float* W1   = (const float*)d_in[4];
    const float* b1   = (const float*)d_in[5];
    const float* W2   = (const float*)d_in[6];
    const float* b2   = (const float*)d_in[7];

    float* newxyz  = (float*)d_out;                         // (4,2048,3)
    float* outFeat = (float*)d_out + NB * NPOINT * 3;       // (4,128,2048)

    char* ws = (char*)d_ws;
    int*      ballIdx = (int*)ws;                           // 1 MiB
    _Float16* W0h = (_Float16*)(ws + 1048576);              // [64][96]
    _Float16* W1h = (_Float16*)(ws + 1048576 + 12288);      // [64][64]
    _Float16* W2h = (_Float16*)(ws + 1048576 + 12288 + 8192);  // [128][64]
    _Float16* fth = (_Float16*)(ws + 1048576 + 36864);      // (4,8192,64) f16

    fps_kernel<<<NB, 512, 0, stream>>>(xyz, newxyz);
    convert_weights<<<72, 256, 0, stream>>>(W0, W1, W2, W0h, W1h, W2h);
    convert_features<<<(NB * CIN * NPTS) / 256, 256, 0, stream>>>(feat, fth);
    ball_query_kernel<<<(NB * NPOINT) / 8, 256, 0, stream>>>(xyz, newxyz, ballIdx);
    mlp_fused_kernel<<<NB * 256, 128, 0, stream>>>(ballIdx, fth, xyz, newxyz,
                                                   W0h, W1h, W2h, b0, b1, b2, outFeat);
}
